// VectorQuantizer_16329465659942
// MI455X (gfx1250) — compile-verified, matrix operand order checked by reading, not run on hardware
//
#include <hip/hip_runtime.h>

// ---------------------------------------------------------------------------
// VQ-VAE vector quantizer for MI455X (gfx1250, wave32, WMMA).
//   z:        (16, 512, 64, 64) fp32   (B, C=e_dim, H, W)
//   codebook: (1024, 512) fp32
// Outputs (concatenated in d_out, float):
//   z_q (B,C,H,W) [33554432] | codebook_loss [1] | idx as float [65536]
//
// Distance argmin uses split-bf16 GEMM: x ~= hi + lo (both bf16),
// dot ~= hi*hi + hi*lo + lo*hi accumulated in f32 via v_wmma_f32_16x16x32_bf16.
// Codebook chunks stream through double-buffered LDS via gfx1250 async
// global->LDS loads (ASYNCcnt); A-operand LDS reads are software-pipelined
// one k-chunk ahead so WMMAs don't stall on ds round-trips.
// ---------------------------------------------------------------------------

typedef __attribute__((ext_vector_type(16))) __bf16          v16bf;
typedef __attribute__((ext_vector_type(8)))  float           v8f;
typedef __attribute__((ext_vector_type(8)))  unsigned short  v8u16;
typedef __attribute__((ext_vector_type(16))) unsigned short  v16u16;
typedef __attribute__((ext_vector_type(4)))  int             v4i;

#if defined(__has_builtin)
#if __has_builtin(__builtin_amdgcn_global_load_async_to_lds_b128)
#define VQ_ASYNC_LDS 1
#endif
#endif

#define AS1 __attribute__((address_space(1)))
#define AS3 __attribute__((address_space(3)))

static constexpr int    E_DIM = 512;
static constexpr int    N_E   = 1024;
static constexpr int    HW    = 4096;                       // 64*64
static constexpr int    NPIX  = 16 * HW;                    // 65536
static constexpr size_t NZ    = (size_t)16 * 512 * 64 * 64; // 33554432

__device__ __forceinline__ unsigned short f2bf(float x) {   // RNE float->bf16
  unsigned u = __float_as_uint(x);
  u += 0x7FFFu + ((u >> 16) & 1u);
  return (unsigned short)(u >> 16);
}
__device__ __forceinline__ float bf2f(unsigned short h) {
  return __uint_as_float(((unsigned)h) << 16);
}

// load one 16-element bf16 A-operand slice (two b128 LDS reads) for this lane
__device__ __forceinline__ v16bf ldA(const unsigned short* p, int c0) {
  v8u16 a = *(const v8u16*)(p + c0);
  v8u16 b = *(const v8u16*)(p + c0 + 16);
  return __builtin_bit_cast(v16bf, __builtin_shufflevector(
      a, b, 0, 1, 2, 3, 4, 5, 6, 7, 8, 9, 10, 11, 12, 13, 14, 15));
}

// ---------------- kernel 1: split codebook into bf16 hi/lo, compute ||c||^2 --
__global__ void __launch_bounds__(256) vq_prep(
    const float* __restrict__ cb,
    unsigned short* __restrict__ cbh,
    unsigned short* __restrict__ cbl,
    float* __restrict__ cnorm,
    float* __restrict__ lossAcc) {
  const int code = blockIdx.x;
  const int t = threadIdx.x;
  float s = 0.f;
#pragma unroll
  for (int j = 0; j < 2; ++j) {
    int c = t + j * 256;
    float x = cb[code * E_DIM + c];
    unsigned short h = f2bf(x);
    cbh[code * E_DIM + c] = h;
    cbl[code * E_DIM + c] = f2bf(x - bf2f(h));
    s += x * x;
  }
#pragma unroll
  for (int o = 16; o >= 1; o >>= 1) s += __shfl_down(s, o, 32);
  __shared__ float red[8];
  if ((t & 31) == 0) red[t >> 5] = s;
  __syncthreads();
  if (t == 0) {
    float tot = 0.f;
#pragma unroll
    for (int i = 0; i < 8; ++i) tot += red[i];
    cnorm[code] = tot;
    if (blockIdx.x == 0) *lossAcc = 0.f;   // zero loss accumulator each launch
  }
}

// ---------------- kernel 2: WMMA distance GEMM + running argmin --------------
// Block: 256 threads (8 waves), owns 128 consecutive pixels; wave w owns 16.
// Each wave keeps its full K=512 z hi/lo bf16 B-operand in VGPRs, then loops
// 64 chunks of 16 codes, codebook staged in double-buffered LDS.
__attribute__((amdgpu_waves_per_eu(2)))          // 2 waves/SIMD -> 512-VGPR cap
__global__ void __launch_bounds__(256) vq_main(
    const float* __restrict__ z,
    const unsigned short* __restrict__ cbh,
    const unsigned short* __restrict__ cbl,
    const float* __restrict__ cnorm,
    int* __restrict__ idxOut) {
  __shared__ union ShMem {
    float zs[128 * 33];                 // z staging (setup phase only)
    unsigned short cb[2][2][16 * 520];  // [buf][hi/lo][16 codes x 512 +8 pad]
  } sh;

  const int t    = threadIdx.x;
  const int lane = t & 31;
  const int wave = t >> 5;
  const int pg   = blockIdx.x;

  const int    bimg   = (pg * 128) >> 12;          // batch index (128 | 4096)
  const int    hwbase = (pg * 128) & (HW - 1);
  const size_t zBase  = (size_t)bimg * E_DIM * HW + hwbase;

  const int prow = wave * 16 + (lane & 15);        // pixel row within block
  const int koff = (lane >> 4) << 3;               // 0 for lanes 0-15, 8 hi

  // ---- stage z (coalesced) -> LDS transpose -> bf16 hi/lo operand registers
  v16bf zh[16], zl[16];
#pragma unroll
  for (int kc = 0; kc < 16; ++kc) {
#pragma unroll
    for (int j = 0; j < 16; ++j) {
      int lin = j * 256 + t;
      int cl = lin >> 7;                 // 0..31 channel within chunk
      int pl = lin & 127;                // pixel within block
      sh.zs[pl * 33 + cl] = z[zBase + (size_t)(kc * 32 + cl) * HW + pl];
    }
    __syncthreads();
    float f[16];
#pragma unroll
    for (int j = 0; j < 8; ++j) {        // WMMA B-operand per-lane K pattern
      f[j]     = sh.zs[prow * 33 + koff + j];
      f[8 + j] = sh.zs[prow * 33 + koff + 16 + j];
    }
    v16u16 uh, ul;
#pragma unroll
    for (int j = 0; j < 16; ++j) {
      unsigned short h = f2bf(f[j]);
      uh[j] = h;
      ul[j] = f2bf(f[j] - bf2f(h));
    }
    zh[kc] = __builtin_bit_cast(v16bf, uh);
    zl[kc] = __builtin_bit_cast(v16bf, ul);
    __syncthreads();
  }

  const uint4* srcH0 = (const uint4*)cbh;
  const uint4* srcL0 = (const uint4*)cbl;

  // fire-and-forget stage of one 16-code chunk (hi+lo, 32 KB) into buffer buf
  auto stage_async = [&](int chunk, int buf) {
    uint4* dH = (uint4*)&sh.cb[buf][0][0];
    uint4* dL = (uint4*)&sh.cb[buf][1][0];
    const uint4* sH = srcH0 + (size_t)chunk * 1024;
    const uint4* sL = srcL0 + (size_t)chunk * 1024;
#pragma unroll
    for (int j = 0; j < 4; ++j) {
      int lin = j * 256 + t;
      int r = lin >> 6, c4 = lin & 63;
#if defined(VQ_ASYNC_LDS)
      __builtin_amdgcn_global_load_async_to_lds_b128(
          (AS1 v4i*)(sH + lin), (AS3 v4i*)(dH + r * 65 + c4), 0, 0);
      __builtin_amdgcn_global_load_async_to_lds_b128(
          (AS1 v4i*)(sL + lin), (AS3 v4i*)(dL + r * 65 + c4), 0, 0);
#else
      dH[r * 65 + c4] = sH[lin];
      dL[r * 65 + c4] = sL[lin];
#endif
    }
  };
  auto wait_async = [&]() {
#if defined(VQ_ASYNC_LDS)
#if __has_builtin(__builtin_amdgcn_s_wait_asynccnt)
    __builtin_amdgcn_s_wait_asynccnt(0);
#else
    asm volatile("s_wait_asynccnt 0x0" ::: "memory");
#endif
#endif
  };

  // ---- stage codebook chunk 0 into buffer 0
  stage_async(0, 0);
  wait_async();
  __syncthreads();

  float minv = 3.0e38f;
  int   mini = 0;
  const int arow = lane & 15;            // codebook row within chunk

#pragma unroll 1
  for (int cc = 0; cc < 64; ++cc) {
    const int cur = cc & 1;

#if defined(VQ_ASYNC_LDS)
    // async prefetch of next chunk into the other buffer (no VGPRs held)
    if (cc + 1 < 64) stage_async(cc + 1, cur ^ 1);
#else
    // register-staged prefetch (512-VGPR budget: no spill)
    uint4 ph[4], plo[4];
    if (cc + 1 < 64) {
      const uint4* sH = srcH0 + (size_t)(cc + 1) * 1024;
      const uint4* sL = srcL0 + (size_t)(cc + 1) * 1024;
#pragma unroll
      for (int j = 0; j < 4; ++j) {
        ph[j]  = sH[j * 256 + t];
        plo[j] = sL[j * 256 + t];
      }
    }
#endif

    // ---- 16 codes x 16 pixels, K=512: 16 k-chunks x 3 split-bf16 WMMAs,
    //      A-operand LDS reads pipelined one k-chunk ahead
    const unsigned short* bh = &sh.cb[cur][0][0] + arow * 520;
    const unsigned short* bl = &sh.cb[cur][1][0] + arow * 520;
    v8f acc = {};
    v16bf ah = ldA(bh, koff);
    v16bf al = ldA(bl, koff);
#pragma unroll
    for (int kc = 0; kc < 16; ++kc) {
      v16bf ahn, aln;
      if (kc < 15) {
        const int c1 = (kc + 1) * 32 + koff;
        ahn = ldA(bh, c1);
        aln = ldA(bl, c1);
      }
      acc = __builtin_amdgcn_wmma_f32_16x16x32_bf16(
          false, ah, false, zh[kc], (short)0, acc, false, false);
      acc = __builtin_amdgcn_wmma_f32_16x16x32_bf16(
          false, ah, false, zl[kc], (short)0, acc, false, false);
      acc = __builtin_amdgcn_wmma_f32_16x16x32_bf16(
          false, al, false, zh[kc], (short)0, acc, false, false);
      ah = ahn;
      al = aln;
    }

    // score = ||c||^2 - 2*dot ; codes strictly increase -> '<' keeps first min
    v8f cn = *(const v8f*)(cnorm + cc * 16 + koff);
#pragma unroll
    for (int v = 0; v < 8; ++v) {
      float s = fmaf(-2.0f, acc[v], cn[v]);
      int code = cc * 16 + koff + v;
      if (s < minv) { minv = s; mini = code; }
    }

#if !defined(VQ_ASYNC_LDS)
    if (cc + 1 < 64) {
      uint4* dH = (uint4*)&sh.cb[cur ^ 1][0][0];
      uint4* dL = (uint4*)&sh.cb[cur ^ 1][1][0];
#pragma unroll
      for (int j = 0; j < 4; ++j) {
        int lin = j * 256 + t;
        int r = lin >> 6, c4 = lin & 63;
        dH[r * 65 + c4] = ph[j];
        dL[r * 65 + c4] = plo[j];
      }
    }
#endif
    wait_async();
    __syncthreads();
  }

  // merge the two lane-halves (codes 0-7 vs 8-15 of each chunk live 16 apart)
  float ov = __shfl_xor(minv, 16, 32);
  int   oi = __shfl_xor(mini, 16, 32);
  if (ov < minv || (ov == minv && oi < mini)) { minv = ov; mini = oi; }
  if (lane < 16) idxOut[pg * 128 + wave * 16 + lane] = mini;
}

// ---------------- kernel 3: gather z_q, write (B,C,H,W), MSE loss ------------
__global__ void __launch_bounds__(256) vq_gather(
    const float* __restrict__ z,
    const float* __restrict__ cb,
    const int* __restrict__ idx,
    float* __restrict__ out,
    float* __restrict__ outIdxF,
    float* __restrict__ lossAcc) {
  const int t = threadIdx.x;
  const int p = blockIdx.x * 256 + t;
  const int id = idx[p];
  outIdxF[p] = (float)id;
  const int bimg = p >> 12;
  const int hw = p & (HW - 1);
  const float4* crow = (const float4*)(cb + (size_t)id * E_DIM);  // contiguous per thread
  const size_t base = (size_t)bimg * E_DIM * HW + hw;
  float sq = 0.f;
#pragma unroll 4
  for (int c4 = 0; c4 < 128; ++c4) {
    float4 q = crow[c4];
    size_t a0 = base + (size_t)(c4 * 4) * HW;
    float z0 = z[a0];           out[a0]          = q.x; float d0 = q.x - z0;
    float z1 = z[a0 + HW];      out[a0 + HW]     = q.y; float d1 = q.y - z1;
    float z2 = z[a0 + 2 * HW];  out[a0 + 2 * HW] = q.z; float d2 = q.z - z2;
    float z3 = z[a0 + 3 * HW];  out[a0 + 3 * HW] = q.w; float d3 = q.w - z3;
    sq += d0 * d0 + d1 * d1 + d2 * d2 + d3 * d3;
  }
#pragma unroll
  for (int o = 16; o >= 1; o >>= 1) sq += __shfl_down(sq, o, 32);
  if ((t & 31) == 0) atomicAdd(lossAcc, sq);
}

// ---------------- kernel 4: finalize loss ------------------------------------
__global__ void vq_final(const float* __restrict__ lossAcc, float* __restrict__ out) {
  // q_latent_loss + BETA * e_latent_loss = (1 + 0.25) * mean((z_q - z)^2)
  out[NZ] = 1.25f * (*lossAcc) / (float)NZ;
}

// ---------------------------------------------------------------------------
extern "C" void kernel_launch(void* const* d_in, const int* in_sizes, int n_in,
                              void* d_out, int out_size, void* d_ws, size_t ws_size,
                              hipStream_t stream) {
  (void)in_sizes; (void)n_in; (void)out_size; (void)ws_size;
  const float* z  = (const float*)d_in[0];
  const float* cb = (const float*)d_in[1];
  float* out = (float*)d_out;

  char* ws = (char*)d_ws;                       // ~2.26 MB of scratch
  unsigned short* cbh   = (unsigned short*)(ws);                 // 1 MB
  unsigned short* cbl   = (unsigned short*)(ws + (1u << 20));    // 1 MB
  float*          cnorm = (float*)(ws + (2u << 20));             // 4 KB (32B aligned)
  int*            idxb  = (int*)(ws + (2u << 20) + 4096);        // 256 KB
  float*          lossA = (float*)(ws + (2u << 20) + 4096 + (size_t)NPIX * 4);

  vq_prep  <<<N_E,        256, 0, stream>>>(cb, cbh, cbl, cnorm, lossA);
  vq_main  <<<NPIX / 128, 256, 0, stream>>>(z, cbh, cbl, cnorm, idxb);
  vq_gather<<<NPIX / 256, 256, 0, stream>>>(z, cb, idxb, out, out + NZ + 1, lossA);
  vq_final <<<1, 1, 0, stream>>>(lossA, out);
}
